// Lfm2ShortConvDecoderLayer_32762010534212
// MI455X (gfx1250) — matrix operand order repaired and checked by reading, not compiled, not run on hardware
//
#include <hip/hip_runtime.h>
#include <hip/hip_bf16.h>

typedef __bf16 bf16_t;
typedef __bf16 v16bf __attribute__((ext_vector_type(16)));
typedef float  v8f   __attribute__((ext_vector_type(8)));

#define TT 32768
#define HH 1024
#define FF 2816
#define EPSV 1e-5f

// LDS tile geometry: K-block = 32 bf16 = 16 uints per row, padded to 20 uints
// (80 B row stride: keeps 16B alignment for b128 ds ops, scatters banks).
#define KB   32
#define ROWU 20

// ---------------- CDNA5 async global->LDS copy (ASYNCcnt tracked) --------
// Each lane DMAs 16 B from its global address directly into LDS, no VGPR
// staging. LDS byte address = low 32 bits of the flat pointer (aperture rule).
__device__ __forceinline__ void async_ld_b128(unsigned int lds_addr, const void* gaddr) {
  asm volatile("global_load_async_to_lds_b128 %0, %1, off"
               :: "v"(lds_addr), "v"(gaddr) : "memory");
}
__device__ __forceinline__ void wait_async0() {
  asm volatile("s_wait_asynccnt 0" ::: "memory");
}
__device__ __forceinline__ unsigned int lds_addr_of(const void* p) {
  return (unsigned int)(unsigned long long)p;
}

// ---------------- fragment loaders (contiguous k per lane half, same order for A and B) ----------------
// A 16x32 bf16: lanes 0-15 row M=lane, K0..15 contiguous;
// lanes 16-31 same rows, K16..31 contiguous (matches fragB).
__device__ __forceinline__ v16bf load_fragA(const unsigned int* s, int row, int kh4) {
  union { uint4 u[2]; v16bf v; } f;
  const unsigned int* p = s + row * ROWU + 2 * kh4;      // kh4 in {0,4}
  f.u[0] = *reinterpret_cast<const uint4*>(p);
  f.u[1] = *reinterpret_cast<const uint4*>(p + 4);
  return f.v;
}
// B 32x16 bf16 (stored as B^T rows in LDS): lane half selects K in [0,16) vs
// [16,32), 8 contiguous uints per lane.
__device__ __forceinline__ v16bf load_fragB(const unsigned int* s, int row, int kh8) {
  union { uint4 u[2]; v16bf v; } f;
  const unsigned int* p = s + row * ROWU + kh8;          // kh8 in {0,8}
  f.u[0] = *reinterpret_cast<const uint4*>(p);
  f.u[1] = *reinterpret_cast<const uint4*>(p + 4);
  return f.v;
}

// fast sigmoid: one v_exp + one v_rcp (no IEEE-div fixup chain)
__device__ __forceinline__ float fast_silu(float g) {
  return g * __builtin_amdgcn_rcpf(1.f + __expf(-g));
}

// ---------------- generic WMMA GEMM ----------------
// C[z] = A[M,K](bf16) * BT[z][N,K]^T (bf16)  (+ bias[z][N]) (+ resid[M,N])
// Block tile 128x128, 8 waves (2M x 4N), wave tile 64x32 => 4x2 frags of 16x16.
// Double-buffered LDS; next tile arrives via async DMA while WMMAs run.
template<bool OUT_BF16>
__global__ __launch_bounds__(256) void gemm_wmma(
    const bf16_t* __restrict__ A,
    const bf16_t* __restrict__ BT,
    const float*  __restrict__ bias,
    const float*  __restrict__ resid,
    void* __restrict__ Cout,
    int M, int N, int K,
    long bSlice, long cSlice, long biasSlice)
{
  const int z = blockIdx.z;
  BT += (long)z * bSlice;
  if (bias) bias += (long)z * biasSlice;

  const int n0 = blockIdx.x * 128;
  const int m0 = blockIdx.y * 128;

  const int tid  = threadIdx.x;
  const int lane = tid & 31;
  const int wave = tid >> 5;
  const int wmo  = (wave >> 2) * 64;
  const int wno  = (wave & 3) * 32;
  const int lr   = lane & 15;
  const int kh4  = ((lane >> 4) & 1) * 4;
  const int kh8  = ((lane >> 4) & 1) * 8;

  __shared__ __align__(16) unsigned int sA[2][128 * ROWU];
  __shared__ __align__(16) unsigned int sB[2][128 * ROWU];

  // DMA layout: 2 threads per row, 32 B (2x b128) each
  const int ldRow  = tid >> 1;
  const int ldHalf = tid & 1;
  const uint4* gA = reinterpret_cast<const uint4*>(A  + (size_t)(m0 + ldRow) * K) + ldHalf * 2;
  const uint4* gB = reinterpret_cast<const uint4*>(BT + (size_t)(n0 + ldRow) * K) + ldHalf * 2;
  const unsigned int ldsA = lds_addr_of(&sA[0][ldRow * ROWU + ldHalf * 8]);
  const unsigned int ldsB = lds_addr_of(&sB[0][ldRow * ROWU + ldHalf * 8]);
  const unsigned int bufStride = 128u * ROWU * 4u;       // sA[0] -> sA[1] bytes

  v8f acc[4][2];
  const v8f vzero = {0.f,0.f,0.f,0.f,0.f,0.f,0.f,0.f};
#pragma unroll
  for (int i = 0; i < 4; ++i)
#pragma unroll
    for (int j = 0; j < 2; ++j) acc[i][j] = vzero;

  const int KT = K / KB;

  // prologue: DMA tile 0 into buffer 0
  async_ld_b128(ldsA,      gA);
  async_ld_b128(ldsA + 16, gA + 1);
  async_ld_b128(ldsB,      gB);
  async_ld_b128(ldsB + 16, gB + 1);
  wait_async0();
  __syncthreads();

  for (int kt = 0; kt < KT; ++kt) {
    const int cur = kt & 1, nxt = cur ^ 1;
    if (kt + 1 < KT) {  // kick off next tile's DMA before computing
      const uint4* pa = gA + (kt + 1) * 4;
      const uint4* pb = gB + (kt + 1) * 4;
      const unsigned int la = ldsA + (unsigned)nxt * bufStride;
      const unsigned int lb = ldsB + (unsigned)nxt * bufStride;
      async_ld_b128(la,      pa);
      async_ld_b128(la + 16, pa + 1);
      async_ld_b128(lb,      pb);
      async_ld_b128(lb + 16, pb + 1);
    }

    v16bf af[4], bfr[2];
#pragma unroll
    for (int fm = 0; fm < 4; ++fm) af[fm]  = load_fragA(sA[cur], wmo + fm * 16 + lr, kh4);
#pragma unroll
    for (int fn = 0; fn < 2; ++fn) bfr[fn] = load_fragB(sB[cur], wno + fn * 16 + lr, kh8);
#pragma unroll
    for (int fm = 0; fm < 4; ++fm)
#pragma unroll
      for (int fn = 0; fn < 2; ++fn)
        acc[fm][fn] = __builtin_amdgcn_wmma_f32_16x16x32_bf16(
            false, af[fm], false, bfr[fn], (short)0, acc[fm][fn], false, false);

    if (kt + 1 < KT) wait_async0();
    __syncthreads();
  }

  // epilogue: lane 0-15 => N=lane, M=v; lane 16-31 => M=v+8
  const int mb = m0 + wmo + ((lane >> 4) << 3);
#pragma unroll
  for (int fm = 0; fm < 4; ++fm) {
#pragma unroll
    for (int fn = 0; fn < 2; ++fn) {
      const int n = n0 + wno + fn * 16 + lr;
      const float bsv = bias ? bias[n] : 0.f;
      const int mrow = mb + fm * 16;
#pragma unroll
      for (int v = 0; v < 8; ++v) {
        const size_t m = (size_t)(mrow + v);
        float val = acc[fm][fn][v] + bsv;
        if (resid) val += resid[m * (size_t)N + n];
        if constexpr (OUT_BF16)
          ((bf16_t*)Cout)[(size_t)z * cSlice + m * (size_t)N + n] = (bf16_t)val;
        else
          ((float*)Cout)[(size_t)z * cSlice + m * (size_t)N + n] = val;
      }
    }
  }
}

// ---------------- fused gate/up WMMA GEMM: act = silu(A*Wg) * (A*Wu) ------
// Block tile 64(M)x128(N) per output matrix; waves 2M x 4N, 2x2 frags each.
__global__ __launch_bounds__(256) void gemm_gateup(
    const bf16_t* __restrict__ A,
    const bf16_t* __restrict__ BTg,
    const bf16_t* __restrict__ BTu,
    bf16_t* __restrict__ act,
    int M, int Nf, int K)
{
  const int n0 = blockIdx.x * 128;
  const int m0 = blockIdx.y * 64;

  const int tid  = threadIdx.x;
  const int lane = tid & 31;
  const int wave = tid >> 5;
  const int wmo  = (wave >> 2) * 32;
  const int wno  = (wave & 3) * 32;
  const int lr   = lane & 15;
  const int kh4  = ((lane >> 4) & 1) * 4;
  const int kh8  = ((lane >> 4) & 1) * 8;

  __shared__ __align__(16) unsigned int sA[2][64 * ROWU];
  __shared__ __align__(16) unsigned int sG[2][128 * ROWU];
  __shared__ __align__(16) unsigned int sU[2][128 * ROWU];

  const int rA = tid >> 2, qA = tid & 3;           // 4 threads per A row, 16 B each
  const int rB = tid >> 1, hB = tid & 1;           // 2 threads per B row, 32 B each
  const uint4* gA = reinterpret_cast<const uint4*>(A   + (size_t)(m0 + rA) * K) + qA;
  const uint4* gG = reinterpret_cast<const uint4*>(BTg + (size_t)(n0 + rB) * K) + hB * 2;
  const uint4* gU = reinterpret_cast<const uint4*>(BTu + (size_t)(n0 + rB) * K) + hB * 2;
  const unsigned int ldsA = lds_addr_of(&sA[0][rA * ROWU + qA * 4]);
  const unsigned int ldsG = lds_addr_of(&sG[0][rB * ROWU + hB * 8]);
  const unsigned int ldsU = lds_addr_of(&sU[0][rB * ROWU + hB * 8]);
  const unsigned int bufA = 64u  * ROWU * 4u;
  const unsigned int bufB = 128u * ROWU * 4u;

  v8f accG[2][2], accU[2][2];
  const v8f vzero = {0.f,0.f,0.f,0.f,0.f,0.f,0.f,0.f};
#pragma unroll
  for (int i = 0; i < 2; ++i)
#pragma unroll
    for (int j = 0; j < 2; ++j) { accG[i][j] = vzero; accU[i][j] = vzero; }

  const int KT = K / KB;

  async_ld_b128(ldsA,      gA);
  async_ld_b128(ldsG,      gG);
  async_ld_b128(ldsG + 16, gG + 1);
  async_ld_b128(ldsU,      gU);
  async_ld_b128(ldsU + 16, gU + 1);
  wait_async0();
  __syncthreads();

  for (int kt = 0; kt < KT; ++kt) {
    const int cur = kt & 1, nxt = cur ^ 1;
    if (kt + 1 < KT) {
      const uint4* pa = gA + (kt + 1) * 4;
      const uint4* pg = gG + (kt + 1) * 4;
      const uint4* pu = gU + (kt + 1) * 4;
      const unsigned int la = ldsA + (unsigned)nxt * bufA;
      const unsigned int lg = ldsG + (unsigned)nxt * bufB;
      const unsigned int lu = ldsU + (unsigned)nxt * bufB;
      async_ld_b128(la,      pa);
      async_ld_b128(lg,      pg);
      async_ld_b128(lg + 16, pg + 1);
      async_ld_b128(lu,      pu);
      async_ld_b128(lu + 16, pu + 1);
    }

    v16bf af[2], bg[2], bu[2];
#pragma unroll
    for (int fm = 0; fm < 2; ++fm) af[fm] = load_fragA(sA[cur], wmo + fm * 16 + lr, kh4);
#pragma unroll
    for (int fn = 0; fn < 2; ++fn) {
      bg[fn] = load_fragB(sG[cur], wno + fn * 16 + lr, kh8);
      bu[fn] = load_fragB(sU[cur], wno + fn * 16 + lr, kh8);
    }
#pragma unroll
    for (int fm = 0; fm < 2; ++fm)
#pragma unroll
      for (int fn = 0; fn < 2; ++fn) {
        accG[fm][fn] = __builtin_amdgcn_wmma_f32_16x16x32_bf16(
            false, af[fm], false, bg[fn], (short)0, accG[fm][fn], false, false);
        accU[fm][fn] = __builtin_amdgcn_wmma_f32_16x16x32_bf16(
            false, af[fm], false, bu[fn], (short)0, accU[fm][fn], false, false);
      }

    if (kt + 1 < KT) wait_async0();
    __syncthreads();
  }

  const int mb = m0 + wmo + ((lane >> 4) << 3);
#pragma unroll
  for (int fm = 0; fm < 2; ++fm)
#pragma unroll
    for (int fn = 0; fn < 2; ++fn) {
      const int n = n0 + wno + fn * 16 + lr;
      const int mrow = mb + fm * 16;
#pragma unroll
      for (int v = 0; v < 8; ++v) {
        const float g = accG[fm][fn][v];
        const float u = accU[fm][fn][v];
        act[(size_t)(mrow + v) * (size_t)Nf + n] = (bf16_t)(fast_silu(g) * u);
      }
    }
}

// ---------------- elementwise / norm / conv / convert kernels ------------

// one block per token; optional add, optional fp32 residual copy, bf16 norm out
__global__ __launch_bounds__(256) void fused_norm(
    const float* __restrict__ a, const float* __restrict__ b,
    const float* __restrict__ g, float* __restrict__ res_out,
    bf16_t* __restrict__ hout)
{
  const int t = blockIdx.x;
  const int tid = threadIdx.x;
  float r[4];
  float ss = 0.f;
#pragma unroll
  for (int i = 0; i < 4; ++i) {
    const int h = tid + i * 256;
    float v = a[(size_t)t * HH + h];
    if (b) v += b[(size_t)t * HH + h];
    r[i] = v;
    ss += v * v;
  }
#pragma unroll
  for (int o = 16; o > 0; o >>= 1) ss += __shfl_xor(ss, o, 32);
  __shared__ float wsum[8];
  if ((tid & 31) == 0) wsum[tid >> 5] = ss;
  __syncthreads();
  float tot = 0.f;
#pragma unroll
  for (int i = 0; i < 8; ++i) tot += wsum[i];
  const float rstd = rsqrtf(tot / (float)HH + EPSV);
#pragma unroll
  for (int i = 0; i < 4; ++i) {
    const int h = tid + i * 256;
    if (res_out) res_out[(size_t)t * HH + h] = r[i];
    hout[(size_t)t * HH + h] = (bf16_t)(r[i] * rstd * g[h]);
  }
}

// Bg *= x (bf16, fp32 math), in place into Bg
__global__ __launch_bounds__(256) void bxmul(bf16_t* __restrict__ Bg,
                                             const bf16_t* __restrict__ x)
{
#pragma unroll
  for (int j = 0; j < 4; ++j) {
    const size_t i = (size_t)blockIdx.x * 1024 + (size_t)j * 256 + threadIdx.x;
    Bg[i] = (bf16_t)((float)Bg[i] * (float)x[i]);
  }
}

// causal K=3 depthwise conv with sequence resets (positions==in-seq index),
// fused with Cg multiply; writes bf16 in place over Cg.
__global__ __launch_bounds__(256) void convmix(
    const bf16_t* __restrict__ Bx, const bf16_t* __restrict__ Cg,
    const float* __restrict__ wconv, const float* __restrict__ bconv,
    const int* __restrict__ positions, bf16_t* __restrict__ out)
{
  const int t = blockIdx.x;
  const int pos = positions[t];
#pragma unroll
  for (int i = 0; i < 4; ++i) {
    const int h = threadIdx.x + i * 256;
    float acc = bconv[h];
#pragma unroll
    for (int s = 0; s < 3; ++s)
      if (pos >= s)
        acc += wconv[h * 3 + (2 - s)] * (float)Bx[(size_t)(t - s) * HH + h];
    out[(size_t)t * HH + h] = (bf16_t)((float)Cg[(size_t)t * HH + h] * acc);
  }
}

// fp32 W[K][N] -> bf16 WT[N][K]
__global__ __launch_bounds__(256) void convT(const float* __restrict__ W,
                                             bf16_t* __restrict__ WT,
                                             int K_, int N_)
{
  const int n = blockIdx.x;
  for (int k = threadIdx.x; k < K_; k += 256)
    WT[(size_t)n * K_ + k] = (bf16_t)W[(size_t)k * N_ + n];
}

// ---------------- launch ----------------
extern "C" void kernel_launch(void* const* d_in, const int* in_sizes, int n_in,
                              void* d_out, int out_size, void* d_ws, size_t ws_size,
                              hipStream_t stream)
{
  (void)in_sizes; (void)n_in; (void)out_size; (void)ws_size;
  const int*   positions = (const int*)d_in[0];
  const float* hs        = (const float*)d_in[1];
  const float* residual  = (const float*)d_in[2];
  const float* g_op      = (const float*)d_in[4];
  const float* g_ffn     = (const float*)d_in[5];
  const float* w_in      = (const float*)d_in[6];
  const float* b_in      = (const float*)d_in[7];
  const float* w_conv    = (const float*)d_in[8];
  const float* b_conv    = (const float*)d_in[9];
  const float* w_out     = (const float*)d_in[10];
  const float* b_out     = (const float*)d_in[11];
  const float* w_gate_up = (const float*)d_in[12];
  const float* w_down    = (const float*)d_in[13];

  // workspace layout (~396 MB):
  //   [hbf 64MB][wt 11.5MB][res 128MB][gem1 192MB]
  //   act (184.5MB) aliases [res..gem1) after both are dead (post-GEMM2).
  char* ws = (char*)d_ws;
  size_t off = 0;
  bf16_t* hbf  = (bf16_t*)(ws + off); off += (size_t)TT * HH * 2;
  bf16_t* wt   = (bf16_t*)(ws + off); off += (size_t)2 * FF * HH * 2;
  const size_t resOff = off;
  float*  resb = (float*)(ws + off);  off += (size_t)TT * HH * 4;
  bf16_t* gem1 = (bf16_t*)(ws + off); off += (size_t)3 * TT * HH * 2;
  bf16_t* act  = (bf16_t*)(ws + resOff);

  float* mlp  = (float*)d_out;
  float* res2 = (float*)d_out + (size_t)TT * HH;

  const dim3 blk(256);

  // 1) res = hs + residual ; h = bf16(rmsnorm(res) * g_op)
  fused_norm<<<TT, blk, 0, stream>>>(hs, residual, g_op, resb, hbf);

  // 2) w_in -> bf16 [3H][H]
  convT<<<3 * HH, blk, 0, stream>>>(w_in, wt, HH, 3 * HH);

  // 3) gem1[z] = h @ w_in[:, zH:(z+1)H] + b_in[zH:(z+1)H], z=0(Bg),1(Cg),2(x)
  gemm_wmma<true><<<dim3(HH / 128, TT / 128, 3), blk, 0, stream>>>(
      hbf, wt, b_in, nullptr, gem1, TT, HH, HH,
      (long)HH * HH, (long)TT * HH, (long)HH);

  // 4) Bx = Bg * x (in place into slice 0)
  bxmul<<<(TT * HH) / 1024, blk, 0, stream>>>(gem1, gem1 + (size_t)2 * TT * HH);

  // 5) CgConv = Cg * causal_conv(Bx) (in place into slice 1)
  convmix<<<TT, blk, 0, stream>>>(gem1, gem1 + (size_t)TT * HH,
                                  w_conv, b_conv, positions,
                                  gem1 + (size_t)TT * HH);

  // 6) w_out -> bf16 [H][H]
  convT<<<HH, blk, 0, stream>>>(w_out, wt, HH, HH);

  // 7) res2 = CgConv @ w_out + b_out + res  (fp32 into d_out)
  gemm_wmma<false><<<dim3(HH / 128, TT / 128, 1), blk, 0, stream>>>(
      gem1 + (size_t)TT * HH, wt, b_out, resb, res2, TT, HH, HH, 0, 0, 0);

  // 8) h2 = bf16(rmsnorm(res2) * g_ffn)
  fused_norm<<<TT, blk, 0, stream>>>(res2, nullptr, g_ffn, nullptr, hbf);

  // 9) w_gate_up -> bf16 [2F][H]
  convT<<<2 * FF, blk, 0, stream>>>(w_gate_up, wt, HH, 2 * FF);

  // 10) act = silu(h2 @ Wgate) * (h2 @ Wup)   (bf16, aliases res/gem1 space)
  gemm_gateup<<<dim3(FF / 128, TT / 64, 1), blk, 0, stream>>>(
      hbf, wt, wt + (size_t)FF * HH, act, TT, FF, HH);

  // 11) w_down -> bf16 [H][F]
  convT<<<HH, blk, 0, stream>>>(w_down, wt, FF, HH);

  // 12) mlp = act @ w_down  (fp32 into d_out)
  gemm_wmma<false><<<dim3(HH / 128, TT / 128, 1), blk, 0, stream>>>(
      act, wt, nullptr, nullptr, mlp, TT, HH, FF, 0, 0, 0);
}
